// Long_Mem_52029233824400
// MI455X (gfx1250) — compile-verified, matrix operand order checked by reading, not run on hardware
//
#include <hip/hip_runtime.h>

#define N_EMBD 2048
#define TT     4096
#define BBATCH 4
#define NH     32
#define HS     64
#define CHL    512
#define MM     (BBATCH * TT)   // 16384 rows

typedef __attribute__((ext_vector_type(16))) _Float16 v16h;
typedef __attribute__((ext_vector_type(8)))  _Float16 v8h;
typedef __attribute__((ext_vector_type(8)))  float    v8f;
typedef __attribute__((ext_vector_type(4)))  unsigned int u32x4;
typedef __attribute__((ext_vector_type(8)))  int i32x8;
typedef __attribute__((ext_vector_type(4)))  int i32x4;

#ifndef __has_builtin
#define __has_builtin(x) 0
#endif
#if __has_builtin(__builtin_amdgcn_tensor_load_to_lds) && \
    __has_builtin(__builtin_amdgcn_s_wait_tensorcnt)
#define HAVE_TDM 1
#else
#define HAVE_TDM 0
#endif

// Build a 16x32 f16 A/B fragment from row-major storage with contiguous K.
// CDNA5 layout: lanes 0-15 hold K {0..7, 16..23}, lanes 16-31 hold
// K {8..15, 24..31}. `p` points at element [row, k0]; the row is already
// selected per-lane by the caller (lane & 15).
static __device__ __forceinline__ v16h frag16(const _Float16* p, int lane) {
  const _Float16* q = p + ((lane & 16) ? 8 : 0);
  v8h lo = *(const v8h*)q;
  v8h hi = *(const v8h*)(q + 16);
  v16h r;
#pragma unroll
  for (int i = 0; i < 8; ++i) { r[i] = lo[i]; r[i + 8] = hi[i]; }
  return r;
}

static __device__ __forceinline__ v8f wmma_f16(v16h a, v16h b, v8f c) {
  return __builtin_amdgcn_wmma_f32_16x16x32_f16(false, a, false, b,
                                                (short)0, c, false, false);
}

// Stage a 128(m) x 32(k) f16 A tile into LDS [m][k].
// TDM path: wave 0 issues tensor_load_to_lds with a 2D descriptor
// (6-arg builtin on this toolchain; groups 2/3 and the extra group zeroed).
// Fallback: all 256 threads copy (two b128 each).
static __device__ __forceinline__ void stageA(_Float16* dst,
                                              const _Float16* A,
                                              int m0, int k0, int M, int K,
                                              int tid) {
#if HAVE_TDM
  if (tid < 32) {
    unsigned long long ga =
        (unsigned long long)(uintptr_t)(A + (size_t)m0 * K + k0);
    unsigned lds = (unsigned)(uintptr_t)dst;
    u32x4 g0 = { 1u,                                   // count = 1
                 lds,                                  // lds_addr
                 (unsigned)(ga & 0xffffffffu),         // global_addr lo
                 (unsigned)(ga >> 32) | 0x80000000u }; // ga hi | type=2
    i32x8 g1 = { (int)0x00010000u,                          // data_size = 2B
                 (int)(((unsigned)K & 0xffffu) << 16),      // tensor_dim0 lo
                 (int)((((unsigned)K >> 16) & 0xffffu) |
                       (((unsigned)M & 0xffffu) << 16)),    // dim0 hi|dim1 lo
                 (int)((((unsigned)M >> 16) & 0xffffu) |
                       (32u << 16)),                        // dim1 hi|tile0=32
                 128,                                       // tile_dim1 = 128
                 K,                                         // dim0_stride lo
                 0, 0 };
    i32x4 z4 = {};
    i32x8 z8 = {};
    __builtin_amdgcn_tensor_load_to_lds(g0, g1, z4, z4, z8, 0);
  }
#else
  const int mrow = tid >> 1;
  const int hp = (tid & 1) * 16;
  const _Float16* src = A + (size_t)(m0 + mrow) * K + k0 + hp;
  v8h a0 = *(const v8h*)src;
  v8h a1 = *(const v8h*)(src + 8);
  *(v8h*)(dst + mrow * 32 + hp) = a0;
  *(v8h*)(dst + mrow * 32 + hp + 8) = a1;
#endif
}

// ---------------------------------------------------------------------------
// f32 -> f16 conversion (weights)
// ---------------------------------------------------------------------------
__global__ __launch_bounds__(256) void f32tof16(const float* __restrict__ s,
                                                _Float16* __restrict__ d,
                                                size_t n) {
  size_t i = (size_t)blockIdx.x * 256 + threadIdx.x;
  if (i < n) d[i] = (_Float16)s[i];
}

// ---------------------------------------------------------------------------
// Time-shift mix: xk/xv/xr/xg in f16
// ---------------------------------------------------------------------------
__global__ __launch_bounds__(256) void timemix(
    const float* __restrict__ x,
    const float* __restrict__ tmk, const float* __restrict__ tmv,
    const float* __restrict__ tmr, const float* __restrict__ tmg,
    _Float16* __restrict__ xk, _Float16* __restrict__ xv,
    _Float16* __restrict__ xr, _Float16* __restrict__ xg) {
  size_t i = (size_t)blockIdx.x * 256 + threadIdx.x;
  if (i >= (size_t)MM * N_EMBD) return;
  int cc = (int)(i & (N_EMBD - 1));
  size_t bt = i >> 11;                 // N_EMBD == 2^11
  int t = (int)(bt & (TT - 1));
  float xc = x[i];
  float xp = t ? x[i - N_EMBD] : 0.0f;
  float mk = tmk[cc], mv = tmv[cc], mr = tmr[cc], mg = tmg[cc];
  xk[i] = (_Float16)(xc * mk + xp * (1.0f - mk));
  xv[i] = (_Float16)(xc * mv + xp * (1.0f - mv));
  xr[i] = (_Float16)(xc * mr + xp * (1.0f - mr));
  xg[i] = (_Float16)(xc * mg + xp * (1.0f - mg));
}

// ---------------------------------------------------------------------------
// Tiled WMMA GEMM: D[MxN] = A[MxK](f16,rm) * B[KxN](f16,rm)
// Block tile 128x128, 8 waves of 2x4 16x16 WMMA tiles, K-step 32.
// A tile staged by TDM (double-buffered LDS); B tile transposed into LDS.
// ---------------------------------------------------------------------------
template <typename OutT>
__global__ __launch_bounds__(256) void gemm_wmma(const _Float16* __restrict__ A,
                                                 const _Float16* __restrict__ Bm,
                                                 OutT* __restrict__ D,
                                                 int M, int N, int K) {
  __shared__ _Float16 At[2][128 * 32];   // [m][k], double-buffered
  __shared__ _Float16 Bt[128 * 32];      // transposed B tile: [n][k]
  const int tid  = threadIdx.x;
  const int lane = tid & 31;
  const int wave = tid >> 5;
  const int wm = wave & 3;               // 4 waves along M (32 rows each)
  const int wn = wave >> 2;              // 2 waves along N (64 cols each)
  const int m0 = blockIdx.x * 128;
  const int n0 = blockIdx.y * 128;
  v8f acc[2][4] = {};

  stageA(&At[0][0], A, m0, 0, M, K, tid);     // prologue

  for (int k0 = 0; k0 < K; k0 += 32) {
    const int buf = (k0 >> 5) & 1;
    {   // cooperative staging of B tile, transposed into LDS
      const int kk = tid >> 3;            // 0..31
      const int ns = (tid & 7) * 16;      // 0..112
      const _Float16* bp = Bm + (size_t)(k0 + kk) * N + n0 + ns;
      v8h b0 = *(const v8h*)bp;
      v8h b1 = *(const v8h*)(bp + 8);
#pragma unroll
      for (int q = 0; q < 8; ++q) {
        Bt[(ns + q) * 32 + kk]     = b0[q];
        Bt[(ns + 8 + q) * 32 + kk] = b1[q];
      }
      if (k0 + 32 < K)
        __builtin_prefetch(bp + (size_t)32 * N, 0, 0);
    }
    if (k0 + 32 < K)                           // async-stage next A tile
      stageA(&At[buf ^ 1][0], A, m0, k0 + 32, M, K, tid);
#if HAVE_TDM
    if (tid < 32) {                            // drain current A tile
      if (k0 + 32 < K) __builtin_amdgcn_s_wait_tensorcnt(1);
      else             __builtin_amdgcn_s_wait_tensorcnt(0);
    }
#endif
    __syncthreads();

    v16h af[2], bf[4];
#pragma unroll
    for (int i = 0; i < 2; ++i)
      af[i] = frag16(&At[buf][(wm * 32 + i * 16 + (lane & 15)) * 32], lane);
#pragma unroll
    for (int j = 0; j < 4; ++j)
      bf[j] = frag16(&Bt[(wn * 64 + j * 16 + (lane & 15)) * 32], lane);
#pragma unroll
    for (int i = 0; i < 2; ++i)
#pragma unroll
      for (int j = 0; j < 4; ++j)
        acc[i][j] = wmma_f16(af[i], bf[j], acc[i][j]);
    __syncthreads();
  }

  const int nl = lane & 15;
  const int mb = (lane & 16) ? 8 : 0;
#pragma unroll
  for (int i = 0; i < 2; ++i)
#pragma unroll
    for (int j = 0; j < 4; ++j)
#pragma unroll
      for (int e = 0; e < 8; ++e) {
        int m = m0 + wm * 32 + i * 16 + mb + e;
        int n = n0 + wn * 64 + j * 16 + nl;
        D[(size_t)m * N + n] = (OutT)acc[i][j][e];
      }
}

// ---------------------------------------------------------------------------
// Chunked RWKV attention: one block per (batch, head), sequential chunks.
// ---------------------------------------------------------------------------
__global__ __launch_bounds__(256) void rwkv_att(const _Float16* __restrict__ r16,
                                                const _Float16* __restrict__ k16,
                                                const _Float16* __restrict__ vv16,
                                                const float* __restrict__ tdecay,
                                                const float* __restrict__ tfaaaa,
                                                const float* __restrict__ state0,
                                                float* __restrict__ attout) {
  __shared__ float    sF[HS * HS];        // running state (f32)
  __shared__ _Float16 s16T[HS * HS];      // transposed f16 snapshot of state
  __shared__ _Float16 kTw[HS * CHL];      // k^T * wk   [s][t]
  __shared__ _Float16 vT [HS * CHL];      // v^T        [s][t]
  __shared__ _Float16 stage[8][16 * 32];  // per-wave att relayout buffer

  const int bh = blockIdx.x;
  const int b = bh >> 5, h = bh & 31;
  const int tid = threadIdx.x, lane = tid & 31, wave = tid >> 5;
  const float w1  = __expf(-__expf(tdecay[h]));
  const float lw  = log2f(w1);
  const float u   = tfaaaa[h];
  const float wsT = exp2f(lw * (float)CHL);

  for (int i = tid; i < HS * HS; i += 256)
    sF[i] = state0[(size_t)bh * (HS * HS) + i];

  const size_t rb0 = (size_t)b * TT;
  for (int c = 0; c < TT / CHL; ++c) {
    __syncthreads();
    const size_t rb = rb0 + (size_t)c * CHL;

    // snapshot state to f16 (transposed for B-fragment reads) + stage tiles
    for (int i = tid; i < HS * HS; i += 256) {
      int a = i >> 6, bc = i & 63;
      s16T[bc * HS + a] = (_Float16)sF[i];
    }
    for (int i = tid; i < CHL * HS; i += 256) {
      int t = i >> 6, s = i & 63;
      float wkt = exp2f(lw * (float)(CHL - 1 - t));
      size_t gi = (rb + t) * (size_t)N_EMBD + h * HS + s;
      kTw[s * CHL + t] = (_Float16)((float)k16[gi] * wkt);
      vT [s * CHL + t] = vv16[gi];
    }
    __syncthreads();

    // out rows: each wave owns i-tiles wave, wave+8, wave+16, wave+24
    for (int it = wave; it < CHL / 16; it += 8) {
      const int i0 = it * 16;
      const int nl = lane & 15;
      const int mb = (lane & 16) ? 8 : 0;
      const _Float16* rrow =
          r16 + (rb + i0 + nl) * (size_t)N_EMBD + h * HS;
      v16h ra0 = frag16(rrow, lane);        // K = s 0..31
      v16h ra1 = frag16(rrow + 32, lane);   // K = s 32..63
      v8f acc[4] = {};

      for (int j0 = 0; j0 < i0 + 16; j0 += 32) {     // causal j-pairs
#pragma unroll
        for (int half = 0; half < 2; ++half) {
          const int jj = j0 + half * 16;
          const _Float16* kcol =
              k16 + (rb + jj + nl) * (size_t)N_EMBD + h * HS;
          v16h kb0 = frag16(kcol, lane);
          v16h kb1 = frag16(kcol + 32, lane);
          v8f p = {};
          p = wmma_f16(ra0, kb0, p);
          p = wmma_f16(ra1, kb1, p);
          // apply decay mask and spill to per-wave staging (f16)
#pragma unroll
          for (int e = 0; e < 8; ++e) {
            int ii = i0 + mb + e;
            int jg = jj + nl;
            float mfac = (jg < ii) ? exp2f(lw * (float)(ii - jg - 1))
                                   : ((jg == ii) ? u : 0.0f);
            stage[wave][(mb + e) * 32 + half * 16 + nl] =
                (_Float16)(p[e] * mfac);
          }
        }
        // reload the masked tile as an A fragment (16x32 over j)
        v16h aA = frag16(&stage[wave][nl * 32], lane);
#pragma unroll
        for (int nb = 0; nb < 4; ++nb) {
          v16h vB = frag16(&vT[(nb * 16 + nl) * CHL + j0], lane);
          acc[nb] = wmma_f16(aA, vB, acc[nb]);
        }
      }

      // cross-chunk term: (r @ s) * w1^(i+1)
      v8f sac[4] = {};
#pragma unroll
      for (int nb = 0; nb < 4; ++nb) {
        v16h sB0 = frag16(&s16T[(nb * 16 + nl) * HS], lane);
        v16h sB1 = frag16(&s16T[(nb * 16 + nl) * HS + 32], lane);
        sac[nb] = wmma_f16(ra0, sB0, sac[nb]);
        sac[nb] = wmma_f16(ra1, sB1, sac[nb]);
      }
#pragma unroll
      for (int nb = 0; nb < 4; ++nb)
#pragma unroll
        for (int e = 0; e < 8; ++e) {
          int ii = i0 + mb + e;
          float wb = exp2f(lw * (float)(ii + 1));
          attout[(rb + ii) * (size_t)N_EMBD + h * HS + nb * 16 + nl] =
              acc[nb][e] + sac[nb][e] * wb;
        }
    }

    // state update: s = w1^T * s + (k^T wk) @ v  (16 tiles, 2 per wave)
    // fragment loads are software-pipelined one K-step ahead of the WMMA.
    {
      const int nl = lane & 15;
      const int mb = (lane & 16) ? 8 : 0;
#pragma unroll
      for (int q = 0; q < 2; ++q) {
        int st = wave * 2 + q;
        int a0 = (st >> 2) * 16, b0 = (st & 3) * 16;
        v8f d = {};
        v16h aK = frag16(&kTw[(a0 + nl) * CHL], lane);
        v16h bV = frag16(&vT [(b0 + nl) * CHL], lane);
        for (int t0 = 0; t0 < CHL; t0 += 32) {
          v16h aN = aK, bN = bV;
          if (t0 + 32 < CHL) {
            aN = frag16(&kTw[(a0 + nl) * CHL + t0 + 32], lane);
            bN = frag16(&vT [(b0 + nl) * CHL + t0 + 32], lane);
          }
          d = wmma_f16(aK, bV, d);
          aK = aN; bV = bN;
        }
#pragma unroll
        for (int e = 0; e < 8; ++e) {
          int a = a0 + mb + e, bc = b0 + nl;
          sF[a * HS + bc] = wsT * sF[a * HS + bc] + d[e];
        }
      }
    }
  }
}

// ---------------------------------------------------------------------------
// GroupNorm(/HEAD_DIV) + SiLU gate -> f16; one 32-lane wave per (row, head)
// ---------------------------------------------------------------------------
__global__ __launch_bounds__(256) void gnorm_gate(const float* __restrict__ att,
                                                  const float* __restrict__ g32,
                                                  const float* __restrict__ gamma,
                                                  const float* __restrict__ beta,
                                                  _Float16* __restrict__ gated) {
  size_t gid = (size_t)blockIdx.x * 8 + (threadIdx.x >> 5);
  int lane = threadIdx.x & 31;
  size_t row = gid >> 5;
  int h = (int)(gid & 31);
  const float* p = att + row * N_EMBD + h * HS;
  float x0 = p[lane] * 0.125f;          // / HEAD_DIV
  float x1 = p[lane + 32] * 0.125f;
  float sum = x0 + x1, sq = x0 * x0 + x1 * x1;
#pragma unroll
  for (int off = 16; off; off >>= 1) {
    sum += __shfl_xor(sum, off, 32);
    sq  += __shfl_xor(sq,  off, 32);
  }
  float mean = sum * (1.0f / 64.0f);
  float var  = sq * (1.0f / 64.0f) - mean * mean;
  float inv  = rsqrtf(var + 1e-5f);
  int c0 = h * HS + lane, c1 = c0 + 32;
  float y0 = (x0 - mean) * inv * gamma[c0] + beta[c0];
  float y1 = (x1 - mean) * inv * gamma[c1] + beta[c1];
  float g0 = g32[row * N_EMBD + c0];  g0 = g0 / (1.0f + __expf(-g0));
  float g1 = g32[row * N_EMBD + c1];  g1 = g1 / (1.0f + __expf(-g1));
  gated[row * N_EMBD + c0] = (_Float16)(y0 * g0);
  gated[row * N_EMBD + c1] = (_Float16)(y1 * g1);
}

// ---------------------------------------------------------------------------
// Launch
// ---------------------------------------------------------------------------
extern "C" void kernel_launch(void* const* d_in, const int* in_sizes, int n_in,
                              void* d_out, int out_size, void* d_ws,
                              size_t ws_size, hipStream_t stream) {
  (void)in_sizes; (void)n_in; (void)out_size; (void)ws_size;
  const float* x      = (const float*)d_in[0];
  const float* W_r    = (const float*)d_in[1];
  const float* W_k    = (const float*)d_in[2];
  const float* W_v    = (const float*)d_in[3];
  const float* W_g    = (const float*)d_in[4];
  const float* W_o    = (const float*)d_in[5];
  const float* gamma  = (const float*)d_in[6];
  const float* beta   = (const float*)d_in[7];
  const float* tmk    = (const float*)d_in[8];
  const float* tmv    = (const float*)d_in[9];
  const float* tmr    = (const float*)d_in[10];
  const float* tmg    = (const float*)d_in[11];
  const float* tdec   = (const float*)d_in[12];
  const float* tfa    = (const float*)d_in[13];
  const float* state0 = (const float*)d_in[14];

  constexpr size_t MC  = (size_t)MM * N_EMBD;        // 33,554,432
  constexpr size_t CCs = (size_t)N_EMBD * N_EMBD;    //  4,194,304
  char* w = (char*)d_ws;
  constexpr size_t o_xr  = 0;
  constexpr size_t o_xk  = o_xr + MC * 2;
  constexpr size_t o_xv  = o_xk + MC * 2;
  constexpr size_t o_xg  = o_xv + MC * 2;
  constexpr size_t o_W   = o_xg + MC * 2;            // 5 f16 weight planes
  constexpr size_t o_r   = o_W + 5 * CCs * 2;
  constexpr size_t o_k   = o_r + MC * 2;
  constexpr size_t o_v   = o_k + MC * 2;
  constexpr size_t o_g   = o_v + MC * 2;             // f32
  constexpr size_t o_att = o_g + MC * 4;             // f32
  constexpr size_t o_gt  = o_att + MC * 4;           // f16

  _Float16* xr   = (_Float16*)(w + o_xr);
  _Float16* xk   = (_Float16*)(w + o_xk);
  _Float16* xv   = (_Float16*)(w + o_xv);
  _Float16* xg   = (_Float16*)(w + o_xg);
  _Float16* W16  = (_Float16*)(w + o_W);
  _Float16* r16  = (_Float16*)(w + o_r);
  _Float16* k16  = (_Float16*)(w + o_k);
  _Float16* v16b = (_Float16*)(w + o_v);
  float*    g32  = (float*)   (w + o_g);
  float*    attf = (float*)   (w + o_att);
  _Float16* gate = (_Float16*)(w + o_gt);

  const int cvtBlocks = (int)(CCs / 256);
  f32tof16<<<cvtBlocks, 256, 0, stream>>>(W_r, W16 + 0 * CCs, CCs);
  f32tof16<<<cvtBlocks, 256, 0, stream>>>(W_k, W16 + 1 * CCs, CCs);
  f32tof16<<<cvtBlocks, 256, 0, stream>>>(W_v, W16 + 2 * CCs, CCs);
  f32tof16<<<cvtBlocks, 256, 0, stream>>>(W_g, W16 + 3 * CCs, CCs);
  f32tof16<<<cvtBlocks, 256, 0, stream>>>(W_o, W16 + 4 * CCs, CCs);

  timemix<<<(int)(MC / 256), 256, 0, stream>>>(x, tmk, tmv, tmr, tmg,
                                               xk, xv, xr, xg);

  dim3 gg(MM / 128, N_EMBD / 128);
  gemm_wmma<_Float16><<<gg, 256, 0, stream>>>(xr, W16 + 0 * CCs, r16,
                                              MM, N_EMBD, N_EMBD);
  gemm_wmma<_Float16><<<gg, 256, 0, stream>>>(xk, W16 + 1 * CCs, k16,
                                              MM, N_EMBD, N_EMBD);
  gemm_wmma<_Float16><<<gg, 256, 0, stream>>>(xv, W16 + 2 * CCs, v16b,
                                              MM, N_EMBD, N_EMBD);
  gemm_wmma<float><<<gg, 256, 0, stream>>>(xg, W16 + 3 * CCs, g32,
                                           MM, N_EMBD, N_EMBD);

  rwkv_att<<<BBATCH * NH, 256, 0, stream>>>(r16, k16, v16b, tdec, tfa,
                                            state0, attf);

  gnorm_gate<<<(MM * NH) / 8, 256, 0, stream>>>(attf, g32, gamma, beta, gate);

  gemm_wmma<float><<<gg, 256, 0, stream>>>(gate, W16 + 4 * CCs, (float*)d_out,
                                           MM, N_EMBD, N_EMBD);
}
